// ColaViTSelfAttn_27358941675842
// MI455X (gfx1250) — matrix operand order repaired and checked by reading, not compiled, not run on hardware
//
#include <hip/hip_runtime.h>
#include <math.h>

// ---- problem dims ----
#define B_   32
#define S_   1026
#define R_   256
#define H_   1024
#define NH_  16
#define HD_  64
#define SPAD 1056   // S padded to multiple of 32 for WMMA tiling

typedef __attribute__((ext_vector_type(16))) __bf16 v16bf;
typedef __attribute__((ext_vector_type(8)))  __bf16 v8bf;
typedef __attribute__((ext_vector_type(4)))  __bf16 v4bf;
typedef __attribute__((ext_vector_type(8)))  float  v8f;
typedef __attribute__((ext_vector_type(4)))  float  v4f;

static __device__ inline v8f wmma_bf16(v16bf a, v16bf b, v8f c) {
  // D = A(16x32 bf16) * B(32x16 bf16) + C(16x16 f32)
  return __builtin_amdgcn_wmma_f32_16x16x32_bf16(false, a, false, b,
                                                 (short)0, c, false, false);
}

static __device__ inline v16bf cat8(v8bf lo, v8bf hi) {
  return __builtin_shufflevector(lo, hi, 0,1,2,3,4,5,6,7,8,9,10,11,12,13,14,15);
}

static __device__ inline v16bf ld16(const __bf16* p) {
  return cat8(*(const v8bf*)p, *(const v8bf*)(p + 8));
}

// A-fragment load: 16x32 bf16 tile, row = lane&15, two 16B segments.
static __device__ inline v16bf ldA(const __bf16* row, int half) {
  return cat8(*(const v8bf*)(row + 16 * half),
              *(const v8bf*)(row + 16 * half + 8));
}

// ============================================================================
// Kernel 0a: transpose+convert a weight matrix W[K,N] f32 -> Wt[N,K] bf16.
// Tiny (<=1 MB each); coalesced reads, scattered 2B writes are fine.
// ============================================================================
__global__ __launch_bounds__(256) void wconv_kernel(
    const float* __restrict__ W, __bf16* __restrict__ Wt, int K, int N)
{
  int gid = blockIdx.x * 256 + threadIdx.x;
  if (gid < K * N) {
    int k = gid / N;
    int n = gid - k * N;
    Wt[(size_t)n * K + k] = (__bf16)W[gid];
  }
}

// ============================================================================
// Kernel 0b: elementwise f32 -> bf16 convert (vector by 4).
// ============================================================================
__global__ __launch_bounds__(256) void xconv_kernel(
    const float* __restrict__ in, __bf16* __restrict__ out, int n4)
{
  int gid = blockIdx.x * 256 + threadIdx.x;
  if (gid < n4) {
    v4f x = *(const v4f*)(in + (size_t)gid * 4);
    v4bf y;
    y[0] = (__bf16)x[0]; y[1] = (__bf16)x[1];
    y[2] = (__bf16)x[2]; y[3] = (__bf16)x[3];
    *(v4bf*)(out + (size_t)gid * 4) = y;
  }
}

// ============================================================================
// Kernel 1: Q/K/V up-projection  Xb[B,S,R](bf16) @ Wt[N=H,K=R](bf16,trans).
// One wave computes a 32x64 tile: 2 M-subtiles x 4 N-subtiles, so each
// K-chunk costs 4 A-loads + 8 B-loads for 8 WMMAs (1.5 b128 per WMMA).
// Q,K stored [B,NH,S_pad,HD]; V stored transposed [B,NH,HD,S_pad].
// Pad rows (s >= S) are zero-filled. grid = (H/64, SPAD/32, B*3)
// ============================================================================
__global__ __launch_bounds__(32) void qkv_proj_kernel(
    const __bf16* __restrict__ Xq, const __bf16* __restrict__ Xk,
    const __bf16* __restrict__ Xv,
    const __bf16* __restrict__ Wtq, const __bf16* __restrict__ Wtk,
    const __bf16* __restrict__ Wtv,
    const float* __restrict__ bq, const float* __restrict__ bk,
    const float* __restrict__ bv,
    __bf16* __restrict__ Qb, __bf16* __restrict__ Kb, __bf16* __restrict__ Vt)
{
  const int lane = threadIdx.x;
  const int l15  = lane & 15;
  const int half = lane >> 4;
  const int n0   = blockIdx.x * 64;
  const int s0   = blockIdx.y * 32;
  const int w    = blockIdx.z % 3;   // 0=Q 1=K 2=V (uniform per block)
  const int b    = blockIdx.z / 3;

  const __bf16* X    = (w == 0) ? Xq  : (w == 1) ? Xk  : Xv;
  const __bf16* Wt   = (w == 0) ? Wtq : (w == 1) ? Wtk : Wtv;
  const float*  bias = (w == 0) ? bq  : (w == 1) ? bk  : bv;

  const __bf16* xrow[2];
#pragma unroll
  for (int mi = 0; mi < 2; ++mi) {
    const int sa = s0 + mi * 16 + l15;
    xrow[mi] = X + ((size_t)b * S_ + (sa < S_ ? sa : 0)) * R_;
  }

  v8f acc[2][4];
#pragma unroll
  for (int mi = 0; mi < 2; ++mi)
#pragma unroll
    for (int t = 0; t < 4; ++t) {
      v8f z = {0.f,0.f,0.f,0.f,0.f,0.f,0.f,0.f};
      acc[mi][t] = z;
    }

  for (int kc = 0; kc < R_ / 32; ++kc) {
    v16bf a[2];
#pragma unroll
    for (int mi = 0; mi < 2; ++mi)
      a[mi] = ldA(xrow[mi] + kc * 32, half);
#pragma unroll
    for (int t = 0; t < 4; ++t) {
      // B 32x16: lane col = n0+16t+l15, elem e -> K = kc*32 + 16*half + e
      const __bf16* wrow = Wt + (size_t)(n0 + t * 16 + l15) * R_
                              + kc * 32 + 16 * half;
      v16bf bfrag = ld16(wrow);
      acc[0][t] = wmma_bf16(a[0], bfrag, acc[0][t]);
      acc[1][t] = wmma_bf16(a[1], bfrag, acc[1][t]);
    }
  }

  const int h = n0 >> 6;   // wave-uniform head index
#pragma unroll
  for (int t = 0; t < 4; ++t) {
    const int ncol = n0 + t * 16 + l15;
    const int d = ncol & 63;
    const float bval = bias[ncol];
#pragma unroll
    for (int mi = 0; mi < 2; ++mi)
#pragma unroll
      for (int r = 0; r < 8; ++r) {          // C/D: row M = r + 8*half
        const int s = s0 + mi * 16 + r + 8 * half;
        const float val = (s < S_) ? (acc[mi][t][r] + bval) : 0.0f;  // pads=0
        if (w == 2) {
          Vt[(((size_t)b * NH_ + h) * HD_ + d) * SPAD + s] = (__bf16)val;
        } else {
          __bf16* dst = (w == 0) ? Qb : Kb;
          dst[(((size_t)b * NH_ + h) * SPAD + s) * HD_ + d] = (__bf16)val;
        }
      }
  }
}

// ============================================================================
// Kernel 2: flash attention, one wave per (b, h, 32-query tile).
// Scores computed TRANSPOSED (St = K_tile x Q^T): the f32 C-layout of St
// (row=key, col=query) coincides with the bf16 A-layout needed for P x V.
// 2 query subtiles share every K/V fragment -> 16 WMMA per 32-key chunk.
// Online softmax stats are lane-local + one shfl_xor(16) per subtile.
// grid = (ceil(S/32), NH, B)
// ============================================================================
__global__ __launch_bounds__(32) void attn_kernel(
    const __bf16* __restrict__ Qb, const __bf16* __restrict__ Kb,
    const __bf16* __restrict__ Vt, __bf16* __restrict__ Cb)
{
  const int lane = threadIdx.x;
  const int l15  = lane & 15;
  const int half = lane >> 4;
  const int q0   = blockIdx.x * 32;
  const int h    = blockIdx.y;
  const int b    = blockIdx.z;

  const __bf16* Qh = Qb + ((size_t)b * NH_ + h) * SPAD * HD_;
  const __bf16* Kh = Kb + ((size_t)b * NH_ + h) * SPAD * HD_;
  const __bf16* Vh = Vt + ((size_t)b * NH_ + h) * HD_  * SPAD;

  // Q^T B-fragments: [qtile][dchunk]; lane col = query, elem e -> d.
  v16bf qt[2][2];
#pragma unroll
  for (int q = 0; q < 2; ++q) {
    const __bf16* qrow = Qh + (size_t)(q0 + q * 16 + l15) * HD_;
#pragma unroll
    for (int dc = 0; dc < 2; ++dc)
      qt[q][dc] = ld16(qrow + dc * 32 + 16 * half);
  }

  v8f ctx[2][4];
#pragma unroll
  for (int q = 0; q < 2; ++q)
#pragma unroll
    for (int t = 0; t < 4; ++t) {
      v8f z = {0.f,0.f,0.f,0.f,0.f,0.f,0.f,0.f};
      ctx[q][t] = z;
    }
  float m[2]    = {-3.0e38f, -3.0e38f};
  float lsum[2] = {0.0f, 0.0f};

  const int NKC = SPAD / 32;   // 33 chunks of 32 keys
  for (int kt = 0; kt < NKC; ++kt) {
    const int k0 = kt * 32;

    // ---- K A-fragments for two 16-key tiles x two 32-d chunks ----
    v16bf ka[2][2];
#pragma unroll
    for (int t = 0; t < 2; ++t) {
      const __bf16* krow = Kh + (size_t)(k0 + t * 16 + l15) * HD_;
#pragma unroll
      for (int dc = 0; dc < 2; ++dc)
        ka[t][dc] = ldA(krow + dc * 32, half);
    }

    // ---- St(key x query): [qtile][keytile] ----
    v8f st[2][2];
#pragma unroll
    for (int q = 0; q < 2; ++q)
#pragma unroll
      for (int t = 0; t < 2; ++t) {
        v8f sacc = {0.f,0.f,0.f,0.f,0.f,0.f,0.f,0.f};
        sacc = wmma_bf16(ka[t][0], qt[q][0], sacc);
        sacc = wmma_bf16(ka[t][1], qt[q][1], sacc);
        st[q][t] = sacc;
      }

    // ---- scale 1/sqrt(64), mask keys >= S, online softmax per qtile ----
    float corr[2];
#pragma unroll
    for (int q = 0; q < 2; ++q) {
      float mloc = -3.0e38f;
#pragma unroll
      for (int t = 0; t < 2; ++t)
#pragma unroll
        for (int r = 0; r < 8; ++r) {
          float v = st[q][t][r] * 0.125f;
          const int key = k0 + t * 16 + r + 8 * half;   // C row M = key
          if (key >= S_) v = -3.0e38f;
          st[q][t][r] = v;
          mloc = fmaxf(mloc, v);
        }
      mloc = fmaxf(mloc, __shfl_xor(mloc, 16, 32));
      const float mnew = fmaxf(m[q], mloc);
      corr[q] = __expf(m[q] - mnew);
      float psum = 0.0f;
#pragma unroll
      for (int t = 0; t < 2; ++t)
#pragma unroll
        for (int r = 0; r < 8; ++r) {
          const float e = __expf(st[q][t][r] - mnew);  // exp in place
          st[q][t][r] = e;
          psum += e;
        }
      psum += __shfl_xor(psum, 16, 32);
      lsum[q] = lsum[q] * corr[q] + psum;
      m[q] = mnew;
    }

    // ---- P A-fragments (lane-local repack) + ctx row rescale ----
    v16bf pa[2];
#pragma unroll
    for (int q = 0; q < 2; ++q) {
#pragma unroll
      for (int e = 0; e < 8; ++e) {
        pa[q][e]     = (__bf16)st[q][0][e];
        pa[q][e + 8] = (__bf16)st[q][1][e];
      }
      float cr[8];
#pragma unroll
      for (int r = 0; r < 8; ++r) cr[r] = __shfl(corr[q], r + 8 * half, 32);
#pragma unroll
      for (int t = 0; t < 4; ++t)
#pragma unroll
        for (int r = 0; r < 8; ++r) ctx[q][t][r] *= cr[r];
    }

    // ---- ctx += P(16x32) x V(32x16): V fragment shared by both qtiles ----
#pragma unroll
    for (int t = 0; t < 4; ++t) {
      const __bf16* vrow = Vh + (size_t)(t * 16 + l15) * SPAD + k0;
      v16bf vb = cat8(*(const v8bf*)(vrow + 8 * half), *(const v8bf*)(vrow + 8 * half + 16));
      ctx[0][t] = wmma_bf16(pa[0], vb, ctx[0][t]);
      ctx[1][t] = wmma_bf16(pa[1], vb, ctx[1][t]);
    }
  }

  // ---- normalize and store ctx as bf16 [B,S,H] ----
#pragma unroll
  for (int q = 0; q < 2; ++q) {
    const float linv = 1.0f / lsum[q];
    float fr[8];
#pragma unroll
    for (int r = 0; r < 8; ++r) fr[r] = __shfl(linv, r + 8 * half, 32);
#pragma unroll
    for (int r = 0; r < 8; ++r) {
      const int s = q0 + q * 16 + r + 8 * half;
      if (s < S_) {
        const size_t base = ((size_t)b * S_ + s) * H_ + (size_t)h * HD_;
#pragma unroll
        for (int t = 0; t < 4; ++t)
          Cb[base + t * 16 + l15] = (__bf16)(ctx[q][t][r] * fr[r]);
      }
    }
  }
}

// ============================================================================
// Kernel 3: out = gelu( ctx[B*S,1024] @ Wo_down[1024,256] ), exact gelu, f32.
// One wave per 32x64 tile (2 M-subtiles x 4 N-subtiles share fragments).
// grid = (R/64, B*S/32)
// ============================================================================
__global__ __launch_bounds__(32) void outproj_kernel(
    const __bf16* __restrict__ Cb, const __bf16* __restrict__ Wto,
    float* __restrict__ out)
{
  const int lane = threadIdx.x;
  const int l15  = lane & 15;
  const int half = lane >> 4;
  const int n0   = blockIdx.x * 64;
  const int m0   = blockIdx.y * 32;

  const __bf16* arow[2];
#pragma unroll
  for (int mi = 0; mi < 2; ++mi)
    arow[mi] = Cb + (size_t)(m0 + mi * 16 + l15) * H_;

  v8f acc[2][4];
#pragma unroll
  for (int mi = 0; mi < 2; ++mi)
#pragma unroll
    for (int t = 0; t < 4; ++t) {
      v8f z = {0.f,0.f,0.f,0.f,0.f,0.f,0.f,0.f};
      acc[mi][t] = z;
    }

  for (int kc = 0; kc < H_ / 32; ++kc) {
    v16bf a[2];
#pragma unroll
    for (int mi = 0; mi < 2; ++mi)
      a[mi] = ldA(arow[mi] + kc * 32, half);
#pragma unroll
    for (int t = 0; t < 4; ++t) {
      const __bf16* wrow = Wto + (size_t)(n0 + t * 16 + l15) * H_
                               + kc * 32 + 16 * half;
      v16bf bfrag = ld16(wrow);
      acc[0][t] = wmma_bf16(a[0], bfrag, acc[0][t]);
      acc[1][t] = wmma_bf16(a[1], bfrag, acc[1][t]);
    }
  }

#pragma unroll
  for (int t = 0; t < 4; ++t) {
    const int ncol = n0 + t * 16 + l15;
#pragma unroll
    for (int mi = 0; mi < 2; ++mi)
#pragma unroll
      for (int r = 0; r < 8; ++r) {
        const float x = acc[mi][t][r];
        const float g = 0.5f * x * (1.0f + erff(x * 0.70710678118f));
        out[(size_t)(m0 + mi * 16 + r + 8 * half) * R_ + ncol] = g;
      }
  }
}

// ============================================================================
extern "C" void kernel_launch(void* const* d_in, const int* in_sizes, int n_in,
                              void* d_out, int out_size, void* d_ws, size_t ws_size,
                              hipStream_t stream) {
  (void)in_sizes; (void)n_in; (void)out_size; (void)ws_size;

  const float* q_low = (const float*)d_in[0];
  const float* k_low = (const float*)d_in[1];
  const float* v_low = (const float*)d_in[2];
  const float* Wq    = (const float*)d_in[3];
  const float* bq    = (const float*)d_in[4];
  const float* Wk    = (const float*)d_in[5];
  const float* bk    = (const float*)d_in[6];
  const float* Wv    = (const float*)d_in[7];
  const float* bv    = (const float*)d_in[8];
  const float* Wo    = (const float*)d_in[9];
  float* out = (float*)d_out;

  // ---- workspace partition (all bf16) ----
  // Qb,Kb,Vt: B*NH*SPAD*HD  (69.2 MB each)
  // Cb:       B*S*H         (67.2 MB)
  // Xq,Xk,Xv: B*S*R         (16.8 MB each)
  // Wtq/k/v, Wto: R*H       (0.5 MB each)     total ~327 MB
  char* ws = (char*)d_ws;
  const size_t qkvSz = (size_t)B_ * NH_ * SPAD * HD_ * sizeof(__bf16);
  const size_t cbSz  = (size_t)B_ * S_ * H_ * sizeof(__bf16);
  const size_t xbSz  = (size_t)B_ * S_ * R_ * sizeof(__bf16);
  const size_t wtSz  = (size_t)R_ * H_ * sizeof(__bf16);
  __bf16* Qb  = (__bf16*)(ws);
  __bf16* Kb  = (__bf16*)(ws + qkvSz);
  __bf16* Vt  = (__bf16*)(ws + 2 * qkvSz);
  __bf16* Cb  = (__bf16*)(ws + 3 * qkvSz);
  __bf16* Xqb = (__bf16*)(ws + 3 * qkvSz + cbSz);
  __bf16* Xkb = (__bf16*)(ws + 3 * qkvSz + cbSz + xbSz);
  __bf16* Xvb = (__bf16*)(ws + 3 * qkvSz + cbSz + 2 * xbSz);
  __bf16* Wtq = (__bf16*)(ws + 3 * qkvSz + cbSz + 3 * xbSz);
  __bf16* Wtk = (__bf16*)(ws + 3 * qkvSz + cbSz + 3 * xbSz + wtSz);
  __bf16* Wtv = (__bf16*)(ws + 3 * qkvSz + cbSz + 3 * xbSz + 2 * wtSz);
  __bf16* Wto = (__bf16*)(ws + 3 * qkvSz + cbSz + 3 * xbSz + 3 * wtSz);

  // ---- stage 0: weight transpose/convert + activation convert ----
  {
    const int wn = R_ * H_;
    const int wgrid = (wn + 255) / 256;
    wconv_kernel<<<wgrid, 256, 0, stream>>>(Wq, Wtq, R_, H_);
    wconv_kernel<<<wgrid, 256, 0, stream>>>(Wk, Wtk, R_, H_);
    wconv_kernel<<<wgrid, 256, 0, stream>>>(Wv, Wtv, R_, H_);
    wconv_kernel<<<wgrid, 256, 0, stream>>>(Wo, Wto, H_, R_);

    const int n4 = (B_ * S_ * R_) / 4;
    const int xgrid = (n4 + 255) / 256;
    xconv_kernel<<<xgrid, 256, 0, stream>>>(q_low, Xqb, n4);
    xconv_kernel<<<xgrid, 256, 0, stream>>>(k_low, Xkb, n4);
    xconv_kernel<<<xgrid, 256, 0, stream>>>(v_low, Xvb, n4);
  }

  // ---- stage 1: QKV projection (32x64 tiles) ----
  qkv_proj_kernel<<<dim3(H_ / 64, SPAD / 32, B_ * 3), 32, 0, stream>>>(
      Xqb, Xkb, Xvb, Wtq, Wtk, Wtv, bq, bk, bv, Qb, Kb, Vt);

  // ---- stage 2: flash attention ----
  attn_kernel<<<dim3((S_ + 31) / 32, NH_, B_), 32, 0, stream>>>(Qb, Kb, Vt, Cb);

  // ---- stage 3: output projection + gelu (32x64 tiles) ----
  outproj_kernel<<<dim3(R_ / 64, (B_ * S_) / 32), 32, 0, stream>>>(Cb, Wto, out);
}
